// DiTMoEBlock_85521388797999
// MI455X (gfx1250) — matrix operand order repaired and checked by reading, not compiled, not run on hardware
//
#include <hip/hip_runtime.h>
#include <hip/hip_bf16.h>

// ---------------------------------------------------------------------------
// DiT + MoE block for gfx1250 (MI455X). bf16 WMMA GEMMs with f32 accumulate.
// Async global->LDS staging (double buffered) + pre-transposed bf16 weights.
// Wave tile 32x64 (8 WMMA / 12 LDS-b128 per K-chunk), block tile 128x128.
// ---------------------------------------------------------------------------

#define D_MODEL   1152
#define N_HEADS   16
#define HEAD_DIM  72
#define N_EXPERTS 8
#define TOPK      2
#define MLP_H     4608          // 4 * D
#define N_TOK     2048          // 8 * 256
#define SEQ       256
#define BATCH     8
#define QKV_N     3456          // 3 * D
#define MOD_N     4608          // 4 * D

#define LDS_STRIDE 40           // padded row stride in halves (80B, 16B-aligned)

typedef __attribute__((ext_vector_type(16))) __bf16 v16bf;
typedef __attribute__((ext_vector_type(8)))  float  v8f;
typedef int v4i_t __attribute__((vector_size(16)));   // exact builtin param type

union Frag { uint4 q[2]; v16bf v; };

#define AS1 __attribute__((address_space(1)))
#define AS3 __attribute__((address_space(3)))

#if defined(__gfx1250__) && \
    __has_builtin(__builtin_amdgcn_global_load_async_to_lds_b128) && \
    __has_builtin(__builtin_amdgcn_s_wait_asynccnt)
#define HAS_ASYNC 1
#else
#define HAS_ASYNC 0
#endif

__device__ __forceinline__ unsigned short f2bf(float f) {
    unsigned u = __float_as_uint(f);
    unsigned r = (u + 0x7FFFu + ((u >> 16) & 1u)) >> 16;   // round-nearest-even
    return (unsigned short)r;
}

__device__ __forceinline__ float gelu_tanh(float x) {
    float x3 = x * x * x;
    return 0.5f * x * (1.0f + tanhf(0.7978845608f * (x + 0.044715f * x3)));
}

// Copy one 16-byte chunk global -> LDS (async on CDNA5, sync fallback).
__device__ __forceinline__ void cp16(const unsigned short* src, unsigned short* dst) {
#if HAS_ASYNC
    __builtin_amdgcn_global_load_async_to_lds_b128(
        (AS1 v4i_t*)src, (AS3 v4i_t*)dst, 0, 0);
#else
    *(uint4*)dst = *(const uint4*)src;
#endif
}

__device__ __forceinline__ void wait_stage() {
#if HAS_ASYNC
    __builtin_amdgcn_s_wait_asynccnt(0);
#endif
    __syncthreads();
}

// ---------------------------------------------------------------------------
// Transpose + f32->bf16 convert:  src (K x N, f32)  ->  dst (N x K, bf16)
// ---------------------------------------------------------------------------
__global__ __launch_bounds__(256) void convT_kernel(
    const float* __restrict__ src, unsigned short* __restrict__ dst,
    int K, int N) {
    __shared__ float t[32][33];
    size_t moff = (size_t)blockIdx.z * (size_t)K * N;
    int k0 = blockIdx.y * 32, n0 = blockIdx.x * 32;
    int tx = threadIdx.x & 31, ty = threadIdx.x >> 5;
#pragma unroll
    for (int i = 0; i < 32; i += 8)
        t[ty + i][tx] = src[moff + (size_t)(k0 + ty + i) * N + n0 + tx];
    __syncthreads();
#pragma unroll
    for (int i = 0; i < 32; i += 8)
        dst[moff + (size_t)(n0 + ty + i) * K + k0 + tx] = f2bf(t[tx][ty + i]);
}

// ---------------------------------------------------------------------------
// mod = silu(c) @ adaln_w + adaln_b     (8 x 4608, K = 1152)
// ---------------------------------------------------------------------------
__global__ void adaln_kernel(const float* __restrict__ c,
                             const float* __restrict__ w,
                             const float* __restrict__ bias,
                             float* __restrict__ mod) {
    __shared__ float sc[D_MODEL];
    int b = blockIdx.y;
    for (int i = threadIdx.x; i < D_MODEL; i += 128) {
        float v = c[b * D_MODEL + i];
        sc[i] = v / (1.0f + __expf(-v));
    }
    __syncthreads();
    int j = blockIdx.x * 128 + threadIdx.x;
    float acc = bias[j];
    for (int k = 0; k < D_MODEL; ++k)
        acc += sc[k] * w[(size_t)k * MOD_N + j];
    mod[b * MOD_N + j] = acc;
}

// ---------------------------------------------------------------------------
// LayerNorm (biased var, eps=1e-6). modulate=1: *(1+scale_msa)+shift_msa.
// ---------------------------------------------------------------------------
__global__ void ln_kernel(const float* __restrict__ X,
                          const float* __restrict__ mod,
                          unsigned short* __restrict__ outbf,
                          int modulate) {
    __shared__ float red[128];
    int n = blockIdx.x;
    int b = n >> 8;
    const float* row = X + (size_t)n * D_MODEL;

    float s = 0.f, s2 = 0.f;
    for (int i = threadIdx.x; i < D_MODEL; i += 128) {
        float v = row[i]; s += v; s2 += v * v;
    }
    red[threadIdx.x] = s; __syncthreads();
    for (int w = 64; w > 0; w >>= 1) {
        if (threadIdx.x < w) red[threadIdx.x] += red[threadIdx.x + w];
        __syncthreads();
    }
    float mean = red[0] * (1.0f / D_MODEL);
    __syncthreads();
    red[threadIdx.x] = s2; __syncthreads();
    for (int w = 64; w > 0; w >>= 1) {
        if (threadIdx.x < w) red[threadIdx.x] += red[threadIdx.x + w];
        __syncthreads();
    }
    float var = red[0] * (1.0f / D_MODEL) - mean * mean;
    float rstd = rsqrtf(var + 1e-6f);

    for (int i = threadIdx.x; i < D_MODEL; i += 128) {
        float v = (row[i] - mean) * rstd;
        if (modulate)
            v = v * (1.0f + mod[b * MOD_N + D_MODEL + i]) + mod[b * MOD_N + i];
        outbf[(size_t)n * D_MODEL + i] = f2bf(v);
    }
}

// ---------------------------------------------------------------------------
// WMMA GEMM: C[M,N] = A[M,K](bf16,row-major) * Bt[N,K](bf16,N-major) + bias.
//   Block tile 128(M) x 128(N); 8 waves = 4(M: 32 rows) x 2(N: 64 cols).
//   Wave computes 2x4 16x16 tiles: 8 WMMA per 32-wide K chunk.
//   Double-buffered LDS staged with async global->LDS b128 (ASYNCcnt).
// MODE 0: qkv   -> Cf  = acc + bias
// MODE 1: proj  -> Cf  = Xres + gate_msa * (acc + bias)
// MODE 2: moe1  -> Cbf = gelu(acc + bias)  (rows gathered via perm, per expert)
// MODE 3: moe2  -> Cf  = acc + bias        (segment rows, per expert)
// ---------------------------------------------------------------------------
template <int MODE>
__global__ __launch_bounds__(256) void gemm_kernel(
    const unsigned short* __restrict__ A, int lda,
    const unsigned short* __restrict__ Bt,   // N x K
    const float* __restrict__ bias,
    float* __restrict__ Cf, unsigned short* __restrict__ Cbf, int ldc,
    int Kdim, int Mfixed,
    const int* __restrict__ cnts, const int* __restrict__ offs,
    const int* __restrict__ perm,
    const float* __restrict__ Xres, const float* __restrict__ mod) {

    __shared__ unsigned short lA[2][128 * LDS_STRIDE];
    __shared__ unsigned short lB[2][128 * LDS_STRIDE];

    int e = blockIdx.z;
    int Mrows = Mfixed;
    const int* rowIdx = nullptr;
    int cOffRows = 0;
    const unsigned short* Ap = A;
    const unsigned short* Bp = Bt;
    const float* biasp = bias;

    if (MODE == 2) {
        Mrows = cnts[e];
        rowIdx = perm + offs[e];
        cOffRows = offs[e];
        Bp = Bt + (size_t)e * (size_t)D_MODEL * MLP_H;
        biasp = bias + e * MLP_H;
    }
    if (MODE == 3) {
        Mrows = cnts[e];
        Ap = A + (size_t)offs[e] * (size_t)lda;
        cOffRows = offs[e];
        Bp = Bt + (size_t)e * (size_t)MLP_H * D_MODEL;
        biasp = bias + e * D_MODEL;
    }

    int mTile = blockIdx.y, nTile = blockIdx.x;
    if (mTile * 128 >= Mrows) return;   // uniform per block

    int tid = threadIdx.x;
    int lane = tid & 31, wave = tid >> 5;
    int wm = wave & 3, wn = wave >> 2;  // wm: 32-row band, wn: 64-col band
    int l15 = lane & 15;
    int hiHalf = (lane >= 16);

    // Per-thread staging: A = 128 rows x 4 chunks (2/thread), same for B.
    const unsigned short* aSrc[2];
    int aDst[2];
#pragma unroll
    for (int c2 = 0; c2 < 2; ++c2) {
        int idx = tid * 2 + c2;
        int r = idx >> 2, seg = idx & 3;
        int gRow = mTile * 128 + r;
        int cl = (gRow < Mrows) ? gRow : (Mrows - 1);
        int ar = (MODE == 2) ? rowIdx[cl] : cl;
        aSrc[c2] = Ap + (size_t)ar * lda + seg * 8;
        aDst[c2] = r * LDS_STRIDE + seg * 8;
    }

    auto stage = [&](int buf, int kk) {
#pragma unroll
        for (int c2 = 0; c2 < 2; ++c2)
            cp16(aSrc[c2] + kk, &lA[buf][aDst[c2]]);
#pragma unroll
        for (int c2 = 0; c2 < 2; ++c2) {
            int idx = tid * 2 + c2;
            int n = idx >> 2, seg = idx & 3;
            cp16(Bp + (size_t)(nTile * 128 + n) * Kdim + kk + seg * 8,
                 &lB[buf][n * LDS_STRIDE + seg * 8]);
        }
    };

    v8f acc[2][4] = {};

    stage(0, 0);
    int cur = 0;
    for (int kk = 0; kk < Kdim; kk += 32) {
        wait_stage();                              // staged buffer ready
        if (kk + 32 < Kdim) stage(cur ^ 1, kk + 32);
        if (kk + 64 < Kdim)                        // prefetch B two chunks out
            __builtin_prefetch(Bp + (size_t)(nTile * 128 + (tid >> 1)) * Kdim
                                  + kk + 64, 0, 0);

        // A fragments: lanes 0-15 K=0..7 & 16..23; lanes 16-31 K=8..15 & 24..31
        Frag a0, a1;
        int aoff0 = (wm * 32 + l15) * LDS_STRIDE + (hiHalf ? 8 : 0);
        int aoff1 = aoff0 + 16 * LDS_STRIDE;
        a0.q[0] = *(const uint4*)&lA[cur][aoff0];
        a0.q[1] = *(const uint4*)&lA[cur][aoff0 + 16];
        a1.q[0] = *(const uint4*)&lA[cur][aoff1];
        a1.q[1] = *(const uint4*)&lA[cur][aoff1 + 16];

        int kbB = hiHalf ? 8 : 0;                  // B: lo K=0..7 & 16..23, hi K=8..15 & 24..31
#pragma unroll
        for (int j = 0; j < 4; ++j) {
            Frag bb;
            int boff = (wn * 64 + j * 16 + l15) * LDS_STRIDE + kbB;
            bb.q[0] = *(const uint4*)&lB[cur][boff];
            bb.q[1] = *(const uint4*)&lB[cur][boff + 16];
            acc[0][j] = __builtin_amdgcn_wmma_f32_16x16x32_bf16(
                false, a0.v, false, bb.v, (short)0, acc[0][j], false, false);
            acc[1][j] = __builtin_amdgcn_wmma_f32_16x16x32_bf16(
                false, a1.v, false, bb.v, (short)0, acc[1][j], false, false);
        }
        cur ^= 1;
        __syncthreads();                           // reads done before rewrite
    }

    // Epilogue: C/D layout — lane col = lane&15, VGPR r -> row r (+8 hi half)
#pragma unroll
    for (int half = 0; half < 2; ++half) {
#pragma unroll
        for (int j = 0; j < 4; ++j) {
#pragma unroll
            for (int r = 0; r < 8; ++r) {
                int rLocal = wm * 32 + half * 16 + r + (hiHalf ? 8 : 0);
                int gRow = mTile * 128 + rLocal;
                int gCol = nTile * 128 + wn * 64 + j * 16 + l15;
                if ((MODE >= 2) && gRow >= Mrows) continue;
                float v = acc[half][j][r] + biasp[gCol];
                if (MODE == 0) {
                    Cf[(size_t)gRow * ldc + gCol] = v;
                } else if (MODE == 1) {
                    int b = gRow >> 8;
                    Cf[(size_t)gRow * D_MODEL + gCol] =
                        Xres[(size_t)gRow * D_MODEL + gCol] +
                        mod[b * MOD_N + 2 * D_MODEL + gCol] * v;
                } else if (MODE == 2) {
                    Cbf[(size_t)(cOffRows + gRow) * ldc + gCol] =
                        f2bf(gelu_tanh(v));
                } else {
                    Cf[(size_t)(cOffRows + gRow) * ldc + gCol] = v;
                }
            }
        }
    }
}

// ---------------------------------------------------------------------------
// Attention: one block per (s, h, b). softmax(q k^T / sqrt(hd)) v   (f32)
// ---------------------------------------------------------------------------
__global__ __launch_bounds__(128) void attn_kernel(
    const float* __restrict__ qkv, unsigned short* __restrict__ o_bf) {
    __shared__ float qv[HEAD_DIM];
    __shared__ float sc[SEQ];
    __shared__ float red[128];

    int s = blockIdx.x, h = blockIdx.y, b = blockIdx.z;
    int t = threadIdx.x;
    const float* base = qkv + (size_t)(b * SEQ) * QKV_N;
    const float* qrow = base + (size_t)s * QKV_N + h * HEAD_DIM;
    if (t < HEAD_DIM) qv[t] = qrow[t];
    __syncthreads();

    const float scale = rsqrtf((float)HEAD_DIM);
    for (int k = t; k < SEQ; k += 128) {
        const float* krow = base + (size_t)k * QKV_N + D_MODEL + h * HEAD_DIM;
        float acc = 0.f;
        for (int d = 0; d < HEAD_DIM; ++d) acc += qv[d] * krow[d];
        sc[k] = acc * scale;
    }
    __syncthreads();

    red[t] = fmaxf(sc[t], sc[t + 128]);
    __syncthreads();
    for (int w = 64; w > 0; w >>= 1) {
        if (t < w) red[t] = fmaxf(red[t], red[t + w]);
        __syncthreads();
    }
    float mx = red[0];
    __syncthreads();
    float e0 = __expf(sc[t] - mx), e1 = __expf(sc[t + 128] - mx);
    sc[t] = e0; sc[t + 128] = e1;
    red[t] = e0 + e1;
    __syncthreads();
    for (int w = 64; w > 0; w >>= 1) {
        if (t < w) red[t] += red[t + w];
        __syncthreads();
    }
    float rs = 1.0f / red[0];
    __syncthreads();

    if (t < HEAD_DIM) {
        float acc = 0.f;
        for (int k = 0; k < SEQ; ++k)
            acc += sc[k] * base[(size_t)k * QKV_N + 2 * D_MODEL + h * HEAD_DIM + t];
        o_bf[(size_t)(b * SEQ + s) * D_MODEL + h * HEAD_DIM + t] = f2bf(acc * rs);
    }
}

// ---------------------------------------------------------------------------
// Router: logits = ln(x2) @ gate_w + gate_b; top-2 softmax; deterministic
// serial slot assignment. Single block of 256 threads.
// ---------------------------------------------------------------------------
__global__ __launch_bounds__(256) void route_kernel(
    const float* __restrict__ x2,
    const float* __restrict__ gw, const float* __restrict__ gb,
    int* __restrict__ tokIdx, float* __restrict__ tokW,
    int* __restrict__ tokSlot, int* __restrict__ perm,
    int* __restrict__ cnts, int* __restrict__ offs) {

    for (int n = threadIdx.x; n < N_TOK; n += 256) {
        const float* row = x2 + (size_t)n * D_MODEL;
        float s = 0.f, s2 = 0.f;
        for (int i = 0; i < D_MODEL; ++i) { float v = row[i]; s += v; s2 += v * v; }
        float mean = s * (1.0f / D_MODEL);
        float rstd = rsqrtf(s2 * (1.0f / D_MODEL) - mean * mean + 1e-6f);
        float lg[N_EXPERTS];
        for (int e = 0; e < N_EXPERTS; ++e) lg[e] = gb[e];
        for (int i = 0; i < D_MODEL; ++i) {
            float v = (row[i] - mean) * rstd;
            for (int e = 0; e < N_EXPERTS; ++e) lg[e] += v * gw[i * N_EXPERTS + e];
        }
        int i0 = 0;
        for (int e = 1; e < N_EXPERTS; ++e) if (lg[e] > lg[i0]) i0 = e;
        int i1 = -1;
        for (int e = 0; e < N_EXPERTS; ++e) {
            if (e == i0) continue;
            if (i1 < 0 || lg[e] > lg[i1]) i1 = e;
        }
        float a1 = __expf(lg[i1] - lg[i0]);
        float inv = 1.0f / (1.0f + a1);
        tokIdx[n * 2] = i0; tokIdx[n * 2 + 1] = i1;
        tokW[n * 2] = inv;  tokW[n * 2 + 1] = a1 * inv;
    }
    __syncthreads();
    if (threadIdx.x == 0) {
        int c[N_EXPERTS];
        for (int e = 0; e < N_EXPERTS; ++e) c[e] = 0;
        for (int n = 0; n < N_TOK; ++n) { c[tokIdx[n*2]]++; c[tokIdx[n*2+1]]++; }
        int o = 0;
        for (int e = 0; e < N_EXPERTS; ++e) { offs[e] = o; cnts[e] = c[e]; o += c[e]; }
        int cur[N_EXPERTS];
        for (int e = 0; e < N_EXPERTS; ++e) cur[e] = offs[e];
        for (int n = 0; n < N_TOK; ++n)
            for (int j = 0; j < TOPK; ++j) {
                int e = tokIdx[n * 2 + j];
                int slot = cur[e]++;
                perm[slot] = n;
                tokSlot[n * 2 + j] = slot;
            }
    }
}

// ---------------------------------------------------------------------------
// Final combine: out = x2 + gate_mlp * (w0*eo[slot0] + w1*eo[slot1])
// ---------------------------------------------------------------------------
__global__ void combine_kernel(const float* __restrict__ x2,
                               const float* __restrict__ mod,
                               const float* __restrict__ eo,
                               const int* __restrict__ tokSlot,
                               const float* __restrict__ tokW,
                               float* __restrict__ out) {
    int n = blockIdx.x;
    int i = blockIdx.y * 128 + threadIdx.x;
    int b = n >> 8;
    int s0 = tokSlot[n * 2], s1 = tokSlot[n * 2 + 1];
    float moe = tokW[n * 2]     * eo[(size_t)s0 * D_MODEL + i] +
                tokW[n * 2 + 1] * eo[(size_t)s1 * D_MODEL + i];
    out[(size_t)n * D_MODEL + i] =
        x2[(size_t)n * D_MODEL + i] + mod[b * MOD_N + 3 * D_MODEL + i] * moe;
}

// ---------------------------------------------------------------------------
// Host-side orchestration
// ---------------------------------------------------------------------------
extern "C" void kernel_launch(void* const* d_in, const int* in_sizes, int n_in,
                              void* d_out, int out_size, void* d_ws, size_t ws_size,
                              hipStream_t stream) {
    const float* x       = (const float*)d_in[0];
    const float* c       = (const float*)d_in[1];
    const float* qkv_w   = (const float*)d_in[2];
    const float* qkv_b   = (const float*)d_in[3];
    const float* proj_w  = (const float*)d_in[4];
    const float* proj_b  = (const float*)d_in[5];
    const float* adaln_w = (const float*)d_in[6];
    const float* adaln_b = (const float*)d_in[7];
    const float* gate_w  = (const float*)d_in[8];
    const float* gate_b  = (const float*)d_in[9];
    const float* w1      = (const float*)d_in[10];
    const float* b1      = (const float*)d_in[11];
    const float* w2      = (const float*)d_in[12];
    const float* b2      = (const float*)d_in[13];
    float* out = (float*)d_out;

    char* wsp = (char*)d_ws;
    auto alloc = [&](size_t bytes) -> void* {
        void* p = (void*)wsp;
        wsp += (bytes + 255) & ~(size_t)255;
        return p;
    };

    float*          mod       = (float*)alloc((size_t)BATCH * MOD_N * 4);
    unsigned short* qkv_wT    = (unsigned short*)alloc((size_t)D_MODEL * QKV_N * 2);
    unsigned short* proj_wT   = (unsigned short*)alloc((size_t)D_MODEL * D_MODEL * 2);
    unsigned short* w1T       = (unsigned short*)alloc((size_t)N_EXPERTS * D_MODEL * MLP_H * 2);
    unsigned short* w2T       = (unsigned short*)alloc((size_t)N_EXPERTS * MLP_H * D_MODEL * 2);
    unsigned short* h_bf      = (unsigned short*)alloc((size_t)N_TOK * D_MODEL * 2);
    float*          qkv_f     = (float*)alloc((size_t)N_TOK * QKV_N * 4);
    unsigned short* o_bf      = (unsigned short*)alloc((size_t)N_TOK * D_MODEL * 2);
    float*          x2        = (float*)alloc((size_t)N_TOK * D_MODEL * 4);
    unsigned short* nx_bf     = (unsigned short*)alloc((size_t)N_TOK * D_MODEL * 2);
    unsigned short* h1_bf     = (unsigned short*)alloc((size_t)N_TOK * TOPK * MLP_H * 2);
    float*          eo        = (float*)alloc((size_t)N_TOK * TOPK * D_MODEL * 4);
    int*            tokIdx    = (int*)alloc((size_t)N_TOK * 2 * 4);
    float*          tokW      = (float*)alloc((size_t)N_TOK * 2 * 4);
    int*            tokSlot   = (int*)alloc((size_t)N_TOK * 2 * 4);
    int*            perm      = (int*)alloc((size_t)N_TOK * TOPK * 4);
    int*            cnts      = (int*)alloc(N_EXPERTS * 4);
    int*            offs      = (int*)alloc(N_EXPERTS * 4);

    // 1) weights -> bf16, transposed to N-major (B side of every GEMM)
    convT_kernel<<<dim3(QKV_N / 32, D_MODEL / 32, 1), 256, 0, stream>>>(
        qkv_w, qkv_wT, D_MODEL, QKV_N);
    convT_kernel<<<dim3(D_MODEL / 32, D_MODEL / 32, 1), 256, 0, stream>>>(
        proj_w, proj_wT, D_MODEL, D_MODEL);
    convT_kernel<<<dim3(MLP_H / 32, D_MODEL / 32, N_EXPERTS), 256, 0, stream>>>(
        w1, w1T, D_MODEL, MLP_H);
    convT_kernel<<<dim3(D_MODEL / 32, MLP_H / 32, N_EXPERTS), 256, 0, stream>>>(
        w2, w2T, MLP_H, D_MODEL);

    // 2) adaLN modulation
    adaln_kernel<<<dim3(MOD_N / 128, BATCH), 128, 0, stream>>>(c, adaln_w, adaln_b, mod);

    // 3) h = ln(x)*(1+scale_msa)+shift_msa -> bf16
    ln_kernel<<<N_TOK, 128, 0, stream>>>(x, mod, h_bf, 1);

    // 4) qkv = h @ qkv_w + qkv_b
    gemm_kernel<0><<<dim3(QKV_N / 128, N_TOK / 128, 1), 256, 0, stream>>>(
        h_bf, D_MODEL, qkv_wT, qkv_b,
        qkv_f, nullptr, QKV_N, D_MODEL, N_TOK,
        nullptr, nullptr, nullptr, nullptr, nullptr);

    // 5) attention -> o_bf
    attn_kernel<<<dim3(SEQ, N_HEADS, BATCH), 128, 0, stream>>>(qkv_f, o_bf);

    // 6) x2 = x + gate_msa * (o @ proj_w + proj_b)
    gemm_kernel<1><<<dim3(D_MODEL / 128, N_TOK / 128, 1), 256, 0, stream>>>(
        o_bf, D_MODEL, proj_wT, proj_b,
        x2, nullptr, D_MODEL, D_MODEL, N_TOK,
        nullptr, nullptr, nullptr, x, mod);

    // 7) nx = ln(x2) -> bf16 ; routing
    ln_kernel<<<N_TOK, 128, 0, stream>>>(x2, nullptr, nx_bf, 0);
    route_kernel<<<1, 256, 0, stream>>>(x2, gate_w, gate_b,
                                        tokIdx, tokW, tokSlot, perm, cnts, offs);

    // 8) expert up-proj: h1 = gelu(nx[perm] @ w1[e] + b1[e]) -> bf16 (compact)
    gemm_kernel<2><<<dim3(MLP_H / 128, N_TOK / 128, N_EXPERTS), 256, 0, stream>>>(
        nx_bf, D_MODEL, w1T, b1,
        nullptr, h1_bf, MLP_H, D_MODEL, 0,
        cnts, offs, perm, nullptr, nullptr);

    // 9) expert down-proj: eo = h1 @ w2[e] + b2[e] -> f32 (compact)
    gemm_kernel<3><<<dim3(D_MODEL / 128, N_TOK / 128, N_EXPERTS), 256, 0, stream>>>(
        h1_bf, MLP_H, w2T, b2,
        eo, nullptr, D_MODEL, MLP_H, 0,
        cnts, offs, perm, nullptr, nullptr);

    // 10) combine
    combine_kernel<<<dim3(N_TOK, D_MODEL / 128), 128, 0, stream>>>(
        x2, mod, eo, tokSlot, tokW, out);
}
